// GlobalPointerLayer_8383776162431
// MI455X (gfx1250) — matrix operand order repaired and checked by reading, not compiled, not run on hardware
//
#include <hip/hip_runtime.h>
#include <hip/hip_bf16.h>
#include <math.h>
#include <stdint.h>

// ---- CDNA5 (gfx1250) wave32 WMMA types -------------------------------------
typedef __attribute__((ext_vector_type(16))) __bf16 v16bf;
typedef __attribute__((ext_vector_type(8)))  __bf16 v8bf;
typedef __attribute__((ext_vector_type(8)))  float  v8f;
typedef __attribute__((ext_vector_type(4)))  int    v4i;
typedef __attribute__((ext_vector_type(8)))  int    v8i;
typedef __attribute__((ext_vector_type(4)))  unsigned int v4u;

#define HEADS     12
#define HEAD_SIZE 64
#define BATCH     16
#define SEQ       512
#define HID       1024
#define NOUT      1536              /* HEADS * HEAD_SIZE * 2 */
#define ROWS      (BATCH * SEQ)     /* 8192 */

// ---- gfx1250 async global->LDS copy (ASYNCcnt-tracked) ---------------------
#if defined(__has_builtin)
#if __has_builtin(__builtin_amdgcn_global_load_async_to_lds_b128)
#define HAVE_ASYNC_LDS 1
#endif
#if __has_builtin(__builtin_amdgcn_tensor_load_to_lds)
#define HAVE_TDM 1
#endif
#endif

typedef __attribute__((address_space(1))) v4i* gptr_v4i;   // global int4*
typedef __attribute__((address_space(3))) v4i* lptr_v4i;   // LDS int4*

__device__ __forceinline__ void async_copy16(const __bf16* g, __bf16* l) {
#ifdef HAVE_ASYNC_LDS
  __builtin_amdgcn_global_load_async_to_lds_b128(
      (gptr_v4i)(g), (lptr_v4i)(l), 0, 0);
#else
  *(v8bf*)l = *(const v8bf*)g;   // fallback: vmem load + ds store
#endif
}

__device__ __forceinline__ void wait_async_zero() {
#if defined(__has_builtin) && __has_builtin(__builtin_amdgcn_s_wait_asynccnt)
  __builtin_amdgcn_s_wait_asynccnt(0);
#else
  asm volatile("s_wait_asynccnt 0x0" ::: "memory");
#endif
}

__device__ __forceinline__ void wait_tensor_zero() {
#if defined(__has_builtin) && __has_builtin(__builtin_amdgcn_s_wait_tensorcnt)
  __builtin_amdgcn_s_wait_tensorcnt(0);
#else
  asm volatile("s_wait_tensorcnt 0x0" ::: "memory");
#endif
}

// ---- Tensor Data Mover: 2D tile (rows x 64 bf16, row stride 768 elems) -----
// D# per CDNA5 ISA. data_size=2B.
__device__ __forceinline__ void tdm_load_2d_bf16(
    const __bf16* gsrc, __bf16* ldst, int rows, int rowlen, int stride_elems) {
#ifdef HAVE_TDM
  const unsigned long long ga = (unsigned long long)(uintptr_t)gsrc;
  const unsigned int       la = (unsigned int)(uintptr_t)ldst;  // LDS byte addr

  v4u g0;
  g0[0] = 1u;                                   // count=1, user descriptor
  g0[1] = la;                                   // lds_addr [63:32]
  g0[2] = (unsigned int)ga;                     // global_addr [95:64]
  g0[3] = (unsigned int)((ga >> 32) & 0x01FFFFFFu) | (2u << 30);  // addr hi | type=2

  v8i g1;
  g1[0] = (1 << 16);                            // data_size=1 (2 bytes); no mask/pad
  g1[1] = (rowlen & 0xFFFF) << 16;              // tensor_dim0[15:0] @ bits[63:48]
  g1[2] = (rows & 0xFFFF) << 16;                // tensor_dim0 hi=0 | tensor_dim1 lo
  g1[3] = (rowlen & 0xFFFF) << 16;              // tensor_dim1 hi=0 | tile_dim0
  g1[4] = (rows & 0xFFFF);                      // tile_dim1 | tile_dim2=0
  g1[5] = stride_elems;                         // tensor_dim0_stride[31:0]
  g1[6] = 0;                                    // stride hi | dim1_stride lo (2D)
  g1[7] = 0;

  const v4i z = {0, 0, 0, 0};                   // groups 2/3 unused (2-D tensor)
#if __clang_major__ >= 23
  const v8i z8 = {0, 0, 0, 0, 0, 0, 0, 0};
  __builtin_amdgcn_tensor_load_to_lds(g0, g1, z, z, z8, 0);
#else
  __builtin_amdgcn_tensor_load_to_lds(g0, g1, z, z, 0);
#endif
#endif
}

// ============================================================================
// Kernel 1: seq = inputs @ W + bias  (8192 x 1536, K=1024), then RoPE,
// store q / k as bf16 into workspace.
// One wave computes a 64(M) x 16(N) strip: 4 WMMA tiles share each B fragment.
// ============================================================================
__global__ __launch_bounds__(256) void proj_rope_kernel(
    const float* __restrict__ inp,   // (8192, 1024)
    const float* __restrict__ W,     // (1024, 1536)
    const float* __restrict__ bias,  // (1536,)
    __bf16* __restrict__ qb,         // (8192, 12, 64) bf16
    __bf16* __restrict__ kb)         // (8192, 12, 64) bf16
{
  const int lane = threadIdx.x & 31;
  const int wave = threadIdx.x >> 5;
  const int l    = lane & 15;
  const int half = lane >> 4;

  const int w       = blockIdx.x * 8 + wave;      // 12288 waves total
  const int tn      = w % (NOUT / 16);            // 0..95
  const int wm      = w / (NOUT / 16);            // 0..127
  const int rowbase = wm * 64;
  const int col     = tn * 16 + l;                // output column (fixed/lane)

  v8f acc[4] = {};

  for (int k0 = 0; k0 < HID; k0 += 32) {
    // ---- B fragment: W[k0 + half*8 + i (+8 for i>=8)][col] (16-bit table K order) --
    const float* wp = W + (size_t)(k0 + half * 8) * NOUT + col;
    v16bf bfrag;
#pragma unroll
    for (int i = 0; i < 16; ++i)
      bfrag[i] = (__bf16)wp[(size_t)(i + (i >= 8 ? 8 : 0)) * NOUT];
    // Prefetch next K-slab of W (global_prefetch_b8, speculative)
    __builtin_prefetch((const void*)(wp + (size_t)32 * NOUT), 0, 0);

#pragma unroll
    for (int mt = 0; mt < 4; ++mt) {
      // ---- A fragment (16-bit A layout): K groups {half*8 .. +7} and +16 --
      const float* ap = inp + (size_t)(rowbase + mt * 16 + l) * HID + k0 + half * 8;
      float4 f0 = *(const float4*)(ap);
      float4 f1 = *(const float4*)(ap + 4);
      float4 f2 = *(const float4*)(ap + 16);
      float4 f3 = *(const float4*)(ap + 20);
      v16bf afrag;
      afrag[0]  = (__bf16)f0.x; afrag[1]  = (__bf16)f0.y;
      afrag[2]  = (__bf16)f0.z; afrag[3]  = (__bf16)f0.w;
      afrag[4]  = (__bf16)f1.x; afrag[5]  = (__bf16)f1.y;
      afrag[6]  = (__bf16)f1.z; afrag[7]  = (__bf16)f1.w;
      afrag[8]  = (__bf16)f2.x; afrag[9]  = (__bf16)f2.y;
      afrag[10] = (__bf16)f2.z; afrag[11] = (__bf16)f2.w;
      afrag[12] = (__bf16)f3.x; afrag[13] = (__bf16)f3.y;
      afrag[14] = (__bf16)f3.z; afrag[15] = (__bf16)f3.w;

      acc[mt] = __builtin_amdgcn_wmma_f32_16x16x32_bf16(
          false, afrag, false, bfrag, (short)0, acc[mt], false, false);
    }
  }

  // ---- Epilogue: +bias, RoPE (pair = adjacent lane), bf16 store ------------
  const float bv    = bias[col];
  const int   r     = col & 127;          // position within head's 2*64 block
  const int   head  = col >> 7;
  const int   d     = r & 63;             // rotary dim
  const bool  is_q  = (r < 64);
  const int   fi    = d >> 1;
  // freq_i = 10000^(-2i/64)
  const float freq  = __expf(-(float)(2 * fi) * (9.210340371976184f / (float)HEAD_SIZE));
  __bf16* base = is_q ? qb : kb;

#pragma unroll
  for (int mt = 0; mt < 4; ++mt) {
#pragma unroll
    for (int j = 0; j < 8; ++j) {
      const int   M    = j + half * 8;              // C layout: lanes 16-31 -> M+8
      const int   grow = rowbase + mt * 16 + M;     // global row = b*512 + m
      const int   pos  = grow & (SEQ - 1);          // sequence position m
      const float ang  = (float)pos * freq;
      const float c    = __cosf(ang);
      const float s    = __sinf(ang);
      float v = acc[mt][j] + bv;
      const float other = __shfl_xor(v, 1, 32);     // partner column (d ^ 1)
      const float outv  = ((col & 1) == 0) ? (v * c - other * s)
                                           : (v * c + other * s);
      base[((size_t)grow * HEADS + head) * HEAD_SIZE + d] = (__bf16)outv;
    }
  }
}

// ============================================================================
// Kernel 2: logits[b,h,m,n] = sum_d q[b,m,h,d] * k[b,n,h,d]  (K = 64)
// + mask -> -inf, causal -1e12, * 1/8.
// One block = (b, h, 128-row m-block).
//   K slab (512x64 bf16, 64KB): one TDM descriptor issued by wave 0
//   Q slab (128x64 bf16, 16KB): async global->LDS (ASYNCcnt)
// then 8 waves x (16 x 512) strips of WMMA fed from ds_load_b128.
// ============================================================================
__global__ __launch_bounds__(256) void qk_logits_kernel(
    const __bf16* __restrict__ qb,
    const __bf16* __restrict__ kb,
    const int*    __restrict__ mask,   // (16, 512)
    float*        __restrict__ out)    // (16, 12, 512, 512)
{
  __shared__ __attribute__((aligned(16))) __bf16 kLDS[SEQ * HEAD_SIZE];  // 64 KB
  __shared__ __attribute__((aligned(16))) __bf16 qLDS[128 * HEAD_SIZE];  // 16 KB

  const int t  = threadIdx.x;
  const int mb = blockIdx.x & 3;                  // 128-row m-block
  const int h  = (blockIdx.x >> 2) % HEADS;
  const int b  = blockIdx.x / (4 * HEADS);

  const int lane = t & 31;
  const int wave = t >> 5;

  // ---- Stage Q slab: 128 rows x 64 bf16 -> 1024 x 16B async chunks ---------
#pragma unroll
  for (int i = 0; i < 4; ++i) {
    const int c = t + i * 256;                    // 0..1023
    const int n = c >> 3, j = c & 7;
    async_copy16(qb + ((size_t)(b * SEQ + mb * 128 + n) * HEADS + h) * HEAD_SIZE + j * 8,
                 &qLDS[n * HEAD_SIZE + j * 8]);
  }

  // ---- Stage K slab: 512 rows x 128B, stride 1536B -> one TDM descriptor ---
#ifdef HAVE_TDM
  if (wave == 0) {
    tdm_load_2d_bf16(kb + ((size_t)(b * SEQ) * HEADS + h) * HEAD_SIZE,
                     kLDS, SEQ, HEAD_SIZE, HEADS * HEAD_SIZE);
    wait_tensor_zero();            // TENSORcnt is per-wave: issuer waits
  }
#else
#pragma unroll
  for (int i = 0; i < 16; ++i) {
    const int c = t + i * 256;                    // 0..4095
    const int n = c >> 3, j = c & 7;
    async_copy16(kb + ((size_t)(b * SEQ + n) * HEADS + h) * HEAD_SIZE + j * 8,
                 &kLDS[n * HEAD_SIZE + j * 8]);
  }
#endif
  wait_async_zero();
  __syncthreads();

  const int l    = lane & 15;
  const int half = lane >> 4;

  // ---- A fragments for this wave's 16-row strip, loaded once from LDS ------
  v16bf afrag[2];
#pragma unroll
  for (int ks = 0; ks < 2; ++ks) {
    const __bf16* ap = &qLDS[(wave * 16 + l) * HEAD_SIZE + ks * 32 + half * 8];
    v8bf a0 = *(const v8bf*)(ap);
    v8bf a1 = *(const v8bf*)(ap + 16);
#pragma unroll
    for (int i = 0; i < 8; ++i) { afrag[ks][i] = a0[i]; afrag[ks][8 + i] = a1[i]; }
  }

  const int  mrow = mb * 128 + wave * 16;         // global m base for this wave
  const int* mk   = mask + b * SEQ;
  int maskm[8];
#pragma unroll
  for (int j = 0; j < 8; ++j) maskm[j] = mk[mrow + j + half * 8];

  float* obase = out + ((size_t)(b * HEADS + h)) * SEQ * SEQ;

  for (int nb = 0; nb < 8; ++nb) {                // 64-column blocks
    v8f acc[4] = {};
#pragma unroll
    for (int ks = 0; ks < 2; ++ks) {
#pragma unroll
      for (int nt = 0; nt < 4; ++nt) {
        const __bf16* bp =
            &kLDS[(nb * 64 + nt * 16 + l) * HEAD_SIZE + ks * 32 + half * 8];
        v8bf b0 = *(const v8bf*)(bp);
        v8bf b1 = *(const v8bf*)(bp + 16);
        v16bf bfrag;
#pragma unroll
        for (int i = 0; i < 8; ++i) { bfrag[i] = b0[i]; bfrag[8 + i] = b1[i]; }

        acc[nt] = __builtin_amdgcn_wmma_f32_16x16x32_bf16(
            false, afrag[ks], false, bfrag, (short)0, acc[nt], false, false);
      }
    }

    // ---- Epilogue: mask, causal tril, scale; coalesced f32 stores ----------
#pragma unroll
    for (int nt = 0; nt < 4; ++nt) {
      const int n     = nb * 64 + nt * 16 + l;
      const int maskn = mk[n];
#pragma unroll
      for (int j = 0; j < 8; ++j) {
        const int m = mrow + j + half * 8;
        float v = acc[nt][j];
        if (!maskn || !maskm[j]) v = -__builtin_inff();
        if (n < m) v -= 1.0e12f;        // tri(k=-1): strictly lower triangle
        v *= 0.125f;                    // 1/sqrt(HEAD_SIZE)
        obase[(size_t)m * SEQ + n] = v;
      }
    }
  }
}

// ============================================================================
extern "C" void kernel_launch(void* const* d_in, const int* in_sizes, int n_in,
                              void* d_out, int out_size, void* d_ws, size_t ws_size,
                              hipStream_t stream) {
  const float* inp  = (const float*)d_in[0];   // (16, 512, 1024) f32
  const int*   mask = (const int*)  d_in[1];   // (16, 512) i32
  const float* W    = (const float*)d_in[2];   // (1024, 1536) f32
  const float* bias = (const float*)d_in[3];   // (1536,) f32
  float*       out  = (float*)d_out;           // (16, 12, 512, 512) f32

  __bf16* qb = (__bf16*)d_ws;                              // 12.6 MB
  __bf16* kb = qb + (size_t)ROWS * HEADS * HEAD_SIZE;      // 12.6 MB

  // 12288 waves = 128 (M/64) * 96 (N/16); 8 waves per block
  proj_rope_kernel<<<1536, 256, 0, stream>>>(inp, W, bias, qb, kb);
  // 768 blocks = 16 (b) * 12 (h) * 4 (m-blocks of 128); 8 waves per block
  qk_logits_kernel<<<768, 256, 0, stream>>>(qb, kb, mask, out);
}
